// MTINN_61847529062621
// MI455X (gfx1250) — compile-verified, matrix operand order checked by reading, not run on hardware
//
#include <hip/hip_runtime.h>

// MTINN multi-task LSTM on MI455X (gfx1250, wave32, WMMA bf16 16x16x32).
// One workgroup per 16-batch tile (16 WGs), 256 threads = 8 waves.
// Weights held in VGPR B-fragments across the 2048-step loop; all recurrent
// state in LDS; gates computed with v_wmma_f32_16x16x32_bf16 (f32 accum).

#define S_LEN 2048
#define NIN   7

typedef __attribute__((ext_vector_type(16))) __bf16 v16bf;
typedef __attribute__((ext_vector_type(8)))  __bf16 v8bf;
typedef __attribute__((ext_vector_type(8)))  float  v8f;

__device__ __forceinline__ float fsig(float x)  { return 1.0f / (1.0f + __expf(-x)); }
__device__ __forceinline__ float ftanh(float x) { return 2.0f * fsig(2.0f * x) - 1.0f; }

__device__ __forceinline__ v8f wmma_bf16(v16bf a, v16bf b, v8f c) {
  // (neg_a, A, neg_b, B, c_mod, C, reuse_a, reuse_b)
  return __builtin_amdgcn_wmma_f32_16x16x32_bf16(false, a, false, b, (short)0, c, false, false);
}

// A fragment (16x32 bf16, M=batch): lane&15 = row M; lanes 0-15 hold K={kh..kh+7, kh+16..},
// per ISA 7.12.2 16-bit A layout. Two 16B LDS loads per lane.
__device__ __forceinline__ v16bf load_afrag(const __bf16* base, int stride, int kbase, int lane) {
  int m  = lane & 15;
  int kh = (lane >> 4) << 3;
  const __bf16* p = base + m * stride + kbase + kh;
  v8bf lo = *(const v8bf*)p;
  v8bf hi = *(const v8bf*)(p + 16);
  v16bf r;
#pragma unroll
  for (int j = 0; j < 8; ++j) { r[j] = lo[j]; r[j + 8] = hi[j]; }
  return r;
}

// C/D tile (16x16 f32): lane&15 = N col, rows M = (lane>>4)*8 + r. Store into
// transposed LDS gate buffer Z[gate][m] -> 8 consecutive floats per lane.
__device__ __forceinline__ void store_ctile(float (*Z)[16], int ntile, v8f acc, int lane) {
  int nn = lane & 15;
  int mb = (lane >> 4) << 3;
#pragma unroll
  for (int r = 0; r < 8; ++r) Z[ntile * 16 + nn][mb + r] = acc[r];
}

// Folded weight element getters (W_hh folded in as extra K columns: leading for shared, trailing for cells).
__device__ __forceinline__ float w_shared(const float* Wih, const float* Whh, int n, int k) {
  if (k < 64) return Whh[n * 64 + k];          // K order: [h_s(64), x(7), pad]
  if (k < 71) return Wih[n * 7 + (k - 64)];
  return 0.0f;
}
__device__ __forceinline__ float w_cell(const float* Wih, const float* Whh, int incols, int n, int k) {
  if (k < incols)      return Wih[n * incols + k];
  if (k < incols + 32) return Whh[n * 32 + (k - incols)];
  return 0.0f;
}

// B fragment (32x16 bf16, K x N): lane&15 = N column, K halves split across lane groups.
__device__ __forceinline__ v16bf gather_shared(const float* Wih, const float* Whh,
                                               int nbase, int kbase, int lane) {
  int n  = nbase + (lane & 15);
  int kh = (lane >> 4) << 3;
  v16bf r;
#pragma unroll
  for (int j = 0; j < 8; ++j) r[j]     = (__bf16)w_shared(Wih, Whh, n, kbase + kh + j);
#pragma unroll
  for (int j = 0; j < 8; ++j) r[j + 8] = (__bf16)w_shared(Wih, Whh, n, kbase + 16 + kh + j);
  return r;
}
__device__ __forceinline__ v16bf gather_cell(const float* Wih, const float* Whh, int incols,
                                             int nbase, int kbase, int lane) {
  int n  = nbase + (lane & 15);
  int kh = (lane >> 4) << 3;
  v16bf r;
#pragma unroll
  for (int j = 0; j < 8; ++j) r[j]     = (__bf16)w_cell(Wih, Whh, incols, n, kbase + kh + j);
#pragma unroll
  for (int j = 0; j < 8; ++j) r[j + 8] = (__bf16)w_cell(Wih, Whh, incols, n, kbase + 16 + kh + j);
  return r;
}

// LSTM elementwise for a 32-wide cell: 16x32 states, 512 items over 256 threads.
__device__ __forceinline__ void cell_ew(const float (*Z)[16], const float* Bg,
                                        float (*C)[32], __bf16* Abase, int stride, int off,
                                        int tid) {
#pragma unroll
  for (int r = 0; r < 2; ++r) {
    int it = tid + r * 256;
    int m = it & 15, j = it >> 4;                 // j in [0,32)
    float iv = Z[j][m]      + Bg[j];
    float fv = Z[32 + j][m] + Bg[32 + j];
    float gv = Z[64 + j][m] + Bg[64 + j];
    float ov = Z[96 + j][m] + Bg[96 + j];
    float c = fsig(fv) * C[m][j] + fsig(iv) * ftanh(gv);
    C[m][j] = c;
    Abase[m * stride + off + j] = (__bf16)(fsig(ov) * ftanh(c));
  }
}

// Two-layer readout head: HC(32)->HQ(16) relu -> OUT(3). Writes output + next cell's A slot.
__device__ __forceinline__ void mlp_head(const float* W1, const float* B1,
                                         const float* W2, const float* B2,
                                         const __bf16* Ain, int astride, int aoff,
                                         float (*Hq)[16],
                                         float* out, int b0, int t, int ooff,
                                         __bf16* Anext, int nstride, int noff, int tid) {
  {
    int m = tid & 15, q = tid >> 4;               // 256 threads = 16x16
    float acc = B1[q];
#pragma unroll
    for (int k = 0; k < 32; ++k) acc += W1[q * 32 + k] * (float)Ain[m * astride + aoff + k];
    Hq[m][q] = fmaxf(acc, 0.0f);
  }
  __syncthreads();
  if (tid < 48) {
    int m = tid / 3, c = tid % 3;
    float acc = B2[c];
#pragma unroll
    for (int q = 0; q < 16; ++q) acc += W2[c * 16 + q] * Hq[m][q];
    out[((size_t)(b0 + m) * S_LEN + t) * 9 + ooff + c] = acc;
    if (Anext) Anext[m * nstride + noff + c] = (__bf16)acc;
  }
  __syncthreads();
}

__global__ __launch_bounds__(256, 1)
void mtinn_kernel(const float* __restrict__ x,
                  const float* __restrict__ Wih_s, const float* __restrict__ Whh_s,
                  const float* __restrict__ bih_s, const float* __restrict__ bhh_s,
                  const float* __restrict__ Wih_a, const float* __restrict__ Whh_a,
                  const float* __restrict__ bih_a, const float* __restrict__ bhh_a,
                  const float* __restrict__ Wih_v, const float* __restrict__ Whh_v,
                  const float* __restrict__ bih_v, const float* __restrict__ bhh_v,
                  const float* __restrict__ Wih_p, const float* __restrict__ Whh_p,
                  const float* __restrict__ bih_p, const float* __restrict__ bhh_p,
                  const float* __restrict__ Wa1, const float* __restrict__ ba1,
                  const float* __restrict__ Wa2, const float* __restrict__ ba2,
                  const float* __restrict__ Wv1, const float* __restrict__ bv1,
                  const float* __restrict__ Wv2, const float* __restrict__ bv2,
                  const float* __restrict__ Wp1, const float* __restrict__ bp1,
                  const float* __restrict__ Wp2, const float* __restrict__ bp2,
                  float* __restrict__ out) {
  // ---- LDS (~59 KB of the WGP's 320 KB) -----------------------------------
  __shared__ __align__(16) __bf16 As[16][96];    // shared A: [h_s(0..63), x(64..70), pad]
  __shared__ __align__(16) __bf16 Aa[16][128];   // att A:    [h_s, x(64..70), h_a(71..102), pad]
  __shared__ __align__(16) __bf16 Av[16][128];   // vel A:    [h_s, d_att(64..66), x(67..73), h_v(74..105), pad]
  __shared__ __align__(16) __bf16 Ap[16][128];   // pos A:    [h_s, d_vel(64..66), h_p(67..98), pad]
  __shared__ __align__(16) float Zs[256][16];    // shared gates, transposed [gate][m]
  __shared__ __align__(16) float Zc[128][16];    // cell gates (reused att/vel/pos)
  __shared__ float Cs[16][64];
  __shared__ float Ca[16][32], Cv[16][32], Cp[16][32];
  __shared__ float Bsg[256], Bag[128], Bvg[128], Bpg[128];   // fused b_ih + b_hh
  __shared__ float Wm1[3][16][32], bm1[3][16], Wm2[3][3][16], bm2[3][3];
  __shared__ float Hq[16][16];

  const int tid  = threadIdx.x;
  const int lane = tid & 31;
  const int wave = tid >> 5;                     // 8 waves
  const int b0   = blockIdx.x * 16;              // batch tile base

  // ---- init LDS: zero states/pads, fuse biases, stage MLP weights ---------
  for (int i = tid; i < 16 * 96; i += 256) (&As[0][0])[i] = (__bf16)0.0f;
  for (int i = tid; i < 16 * 128; i += 256) {
    (&Aa[0][0])[i] = (__bf16)0.0f; (&Av[0][0])[i] = (__bf16)0.0f; (&Ap[0][0])[i] = (__bf16)0.0f;
  }
  for (int i = tid; i < 16 * 64; i += 256) (&Cs[0][0])[i] = 0.0f;
  for (int i = tid; i < 16 * 32; i += 256) {
    (&Ca[0][0])[i] = 0.0f; (&Cv[0][0])[i] = 0.0f; (&Cp[0][0])[i] = 0.0f;
  }
  if (tid < 256) Bsg[tid] = bih_s[tid] + bhh_s[tid];
  if (tid < 128) {
    Bag[tid] = bih_a[tid] + bhh_a[tid];
    Bvg[tid] = bih_v[tid] + bhh_v[tid];
    Bpg[tid] = bih_p[tid] + bhh_p[tid];
  }
  for (int i = tid; i < 512; i += 256) {
    (&Wm1[0][0][0])[i] = Wa1[i]; (&Wm1[1][0][0])[i] = Wv1[i]; (&Wm1[2][0][0])[i] = Wp1[i];
  }
  if (tid < 16) { bm1[0][tid] = ba1[tid]; bm1[1][tid] = bv1[tid]; bm1[2][tid] = bp1[tid]; }
  if (tid < 48) { (&Wm2[0][0][0])[tid] = Wa2[tid]; (&Wm2[1][0][0])[tid] = Wv2[tid]; (&Wm2[2][0][0])[tid] = Wp2[tid]; }
  if (tid < 3)  { bm2[0][tid] = ba2[tid]; bm2[1][tid] = bv2[tid]; bm2[2][tid] = bp2[tid]; }

  // ---- pin weight B-fragments in VGPRs for the whole sequence -------------
  v16bf WsB[2][3];                               // shared: 2 N-tiles/wave x 3 K-chunks
#pragma unroll
  for (int p = 0; p < 2; ++p)
#pragma unroll
    for (int kc = 0; kc < 3; ++kc)
      WsB[p][kc] = gather_shared(Wih_s, Whh_s, (wave + 8 * p) * 16, kc * 32, lane);
  v16bf WaB[4], WvB[4], WpB[4];                  // cells: 1 N-tile/wave x 4 K-chunks
#pragma unroll
  for (int kc = 0; kc < 4; ++kc) {
    WaB[kc] = gather_cell(Wih_a, Whh_a, 71, wave * 16, kc * 32, lane);
    WvB[kc] = gather_cell(Wih_v, Whh_v, 74, wave * 16, kc * 32, lane);
    WpB[kc] = gather_cell(Wih_p, Whh_p, 67, wave * 16, kc * 32, lane);
  }
  __syncthreads();

  // ---- recurrence over S=2048 steps ---------------------------------------
  for (int t = 0; t < S_LEN; ++t) {
    // stage x_t into the A buffers (bf16)
    if (tid < 16 * NIN) {
      int m = tid / NIN, j = tid % NIN;
      __bf16 xb = (__bf16)x[(size_t)(b0 + m) * S_LEN * NIN + (size_t)t * NIN + j];
      As[m][64 + j] = xb; Aa[m][64 + j] = xb; Av[m][67 + j] = xb;
    }
    if (tid < 16 && t + 1 < S_LEN)
      __builtin_prefetch(&x[(size_t)(b0 + tid) * S_LEN * NIN + (size_t)(t + 1) * NIN], 0, 0);
    __syncthreads();

    // shared LSTM gates: [16 x 96] @ [96 x 256], 6 WMMA/wave
    {
      v16bf a0 = load_afrag(&As[0][0], 96, 0,  lane);
      v16bf a1 = load_afrag(&As[0][0], 96, 32, lane);
      v16bf a2 = load_afrag(&As[0][0], 96, 64, lane);
      v8f acc0, acc1;
#pragma unroll
      for (int r = 0; r < 8; ++r) { acc0[r] = 0.0f; acc1[r] = 0.0f; }
      acc0 = wmma_bf16(a0, WsB[0][0], acc0);
      acc0 = wmma_bf16(a1, WsB[0][1], acc0);
      acc0 = wmma_bf16(a2, WsB[0][2], acc0);
      acc1 = wmma_bf16(a0, WsB[1][0], acc1);
      acc1 = wmma_bf16(a1, WsB[1][1], acc1);
      acc1 = wmma_bf16(a2, WsB[1][2], acc1);
      store_ctile(Zs, wave,     acc0, lane);
      store_ctile(Zs, wave + 8, acc1, lane);
    }
    __syncthreads();

    // shared elementwise: 16x64 states, h_s broadcast (bf16) into all 4 A buffers
#pragma unroll
    for (int r = 0; r < 4; ++r) {
      int it = tid + r * 256;
      int m = it & 15, j = it >> 4;              // j in [0,64)
      float iv = Zs[j][m]       + Bsg[j];
      float fv = Zs[64 + j][m]  + Bsg[64 + j];
      float gv = Zs[128 + j][m] + Bsg[128 + j];
      float ov = Zs[192 + j][m] + Bsg[192 + j];
      float c = fsig(fv) * Cs[m][j] + fsig(iv) * ftanh(gv);
      Cs[m][j] = c;
      __bf16 hb = (__bf16)(fsig(ov) * ftanh(c));
      As[m][j] = hb; Aa[m][j] = hb; Av[m][j] = hb; Ap[m][j] = hb;
    }
    __syncthreads();

    // attitude cell: [16x128] @ [128x128], 4 WMMA/wave
    {
      v8f acc;
#pragma unroll
      for (int r = 0; r < 8; ++r) acc[r] = 0.0f;
#pragma unroll
      for (int kc = 0; kc < 4; ++kc)
        acc = wmma_bf16(load_afrag(&Aa[0][0], 128, kc * 32, lane), WaB[kc], acc);
      store_ctile(Zc, wave, acc, lane);
    }
    __syncthreads();
    cell_ew(Zc, Bag, Ca, &Aa[0][0], 128, 71, tid);
    __syncthreads();
    mlp_head(&Wm1[0][0][0], bm1[0], &Wm2[0][0][0], bm2[0],
             &Aa[0][0], 128, 71, Hq, out, b0, t, 0, &Av[0][0], 128, 64, tid);

    // velocity cell
    {
      v8f acc;
#pragma unroll
      for (int r = 0; r < 8; ++r) acc[r] = 0.0f;
#pragma unroll
      for (int kc = 0; kc < 4; ++kc)
        acc = wmma_bf16(load_afrag(&Av[0][0], 128, kc * 32, lane), WvB[kc], acc);
      store_ctile(Zc, wave, acc, lane);
    }
    __syncthreads();
    cell_ew(Zc, Bvg, Cv, &Av[0][0], 128, 74, tid);
    __syncthreads();
    mlp_head(&Wm1[1][0][0], bm1[1], &Wm2[1][0][0], bm2[1],
             &Av[0][0], 128, 74, Hq, out, b0, t, 3, &Ap[0][0], 128, 64, tid);

    // position cell
    {
      v8f acc;
#pragma unroll
      for (int r = 0; r < 8; ++r) acc[r] = 0.0f;
#pragma unroll
      for (int kc = 0; kc < 4; ++kc)
        acc = wmma_bf16(load_afrag(&Ap[0][0], 128, kc * 32, lane), WpB[kc], acc);
      store_ctile(Zc, wave, acc, lane);
    }
    __syncthreads();
    cell_ew(Zc, Bpg, Cp, &Ap[0][0], 128, 67, tid);
    __syncthreads();
    mlp_head(&Wm1[2][0][0], bm1[2], &Wm2[2][0][0], bm2[2],
             &Ap[0][0], 128, 67, Hq, out, b0, t, 6, (__bf16*)nullptr, 0, 0, tid);
  }
}

extern "C" void kernel_launch(void* const* d_in, const int* in_sizes, int n_in,
                              void* d_out, int out_size, void* d_ws, size_t ws_size,
                              hipStream_t stream) {
  (void)in_sizes; (void)n_in; (void)out_size; (void)d_ws; (void)ws_size;
  const float* p[29];
  for (int i = 0; i < 29; ++i) p[i] = (const float*)d_in[i];
  dim3 grid(16), block(256);
  mtinn_kernel<<<grid, block, 0, stream>>>(
      p[0],
      p[1],  p[2],  p[3],  p[4],
      p[5],  p[6],  p[7],  p[8],
      p[9],  p[10], p[11], p[12],
      p[13], p[14], p[15], p[16],
      p[17], p[18], p[19], p[20],
      p[21], p[22], p[23], p[24],
      p[25], p[26], p[27], p[28],
      (float*)d_out);
}
